// CTC_16793322127773
// MI455X (gfx1250) — matrix operand order repaired and checked by reading, not compiled, not run on hardware
//
#include <hip/hip_runtime.h>
#include <hip/hip_bf16.h>
#include <stdint.h>
#include <stddef.h>

// ---------------- CDNA5 WMMA types ----------------
typedef __attribute__((ext_vector_type(16))) __bf16 v16bf;
typedef __attribute__((ext_vector_type(8)))  float  v8f;

#define NEGF (-1e30f)

// problem sizes (from reference setup_inputs)
#define Bsz   32
#define Tsz   800
#define Dsz   512           // K
#define Vsz   5000
#define Lsz   100
#define Mrows (Bsz*Tsz)     // 25600
#define NPAD  5120          // V padded to multiple of 128
#define KB32  16            // Dsz/32
#define NCT   40            // NPAD/128 column tiles
#define NLAB  112           // 101 label tokens padded to 7*16
#define SEXT  201           // 2*L+1

// fp32 -> bf16 round-to-nearest-even
static __device__ __forceinline__ unsigned short f2bf(float f) {
  unsigned u = __builtin_bit_cast(unsigned, f);
  unsigned r = u + 0x7FFFu + ((u >> 16) & 1u);
  return (unsigned short)(r >> 16);
}

// ---------------------------------------------------------------------------
// K-zero: init the scalar output (loss accumulator)
// ---------------------------------------------------------------------------
__global__ void k_zero(float* out) {
  if (threadIdx.x == 0 && blockIdx.x == 0) out[0] = 0.0f;
}

// ---------------------------------------------------------------------------
// K0: convert hs_pad [25600 x 512] fp32 -> bf16 in A-fragment layout.
// A frag per lane (16 bf16): half h=lane>>4; element j -> k = h*16 + j
// (contig).  Storage: ((m*KB32 + kb)*2 + h)*16.
// ---------------------------------------------------------------------------
__global__ void k_convA(const float* __restrict__ hs, unsigned short* __restrict__ A) {
  int idx = blockIdx.x * 256 + threadIdx.x;          // fragment id
  if (idx >= Mrows * KB32 * 2) return;
  int h  = idx & 1;
  int kb = (idx >> 1) & (KB32 - 1);
  int m  = idx >> 5;                                 // KB32*2 == 32
  const float* src = hs + (size_t)m * Dsz + kb * 32 + h * 16;
  unsigned short* dst = A + (size_t)idx * 16;
#pragma unroll
  for (int j = 0; j < 8; ++j) dst[j]     = f2bf(src[j]);
#pragma unroll
  for (int j = 0; j < 8; ++j) dst[8 + j] = f2bf(src[8 + j]);
}

// ---------------------------------------------------------------------------
// K1: convert W [5000 x 512] fp32 -> bf16 in B-fragment layout.
// B frag per lane (16 bf16): column n = lane&15, k = (lane>>4)*16 + j (contig).
// Storage: ((kb16*NPAD + v)*16 + j), kb16 = k/16 in 0..31.  v>=5000 -> 0.
// ---------------------------------------------------------------------------
__global__ void k_convW(const float* __restrict__ W, unsigned short* __restrict__ Bw) {
  int idx = blockIdx.x * 256 + threadIdx.x;          // idx = kb16*NPAD + v
  if (idx >= 32 * NPAD) return;
  int v    = idx % NPAD;
  int kb16 = idx / NPAD;
  unsigned short* dst = Bw + (size_t)idx * 16;
  if (v < Vsz) {
    const float* src = W + (size_t)v * Dsz + kb16 * 16;
#pragma unroll
    for (int j = 0; j < 16; ++j) dst[j] = f2bf(src[j]);
  } else {
#pragma unroll
    for (int j = 0; j < 16; ++j) dst[j] = 0;
  }
}

// ---------------------------------------------------------------------------
// K2: main GEMM (bf16 WMMA) fused with per-tile softmax partials.
// Block = 256 thr = 8 waves; tile 128x128; wave -> 32 rows x 64 cols
// (2x4 16x16 WMMA accumulators). K loop: 16 steps of 32.
// Bias preloaded branchlessly before the K loop (latency hidden under WMMAs).
// Epilogue: per-row (max, sum exp) over this column tile -> pmax/psum.
// ---------------------------------------------------------------------------
__global__ __launch_bounds__(256) void k_gemm(
    const unsigned short* __restrict__ A, const unsigned short* __restrict__ Bw,
    const float* __restrict__ bias, float* __restrict__ pmax, float* __restrict__ psum) {
  const int tid  = threadIdx.x;
  const int wave = tid >> 5, lane = tid & 31;
  const int half = lane >> 4, ln = lane & 15;
  const int wr = wave & 3, wc = wave >> 2;
  const int tileM = blockIdx.x * 128, tileN = blockIdx.y * 128;

  // branchless bias preload for this lane's 4 columns (clamped index)
  float bcol[4];
  bool  cvalid[4];
#pragma unroll
  for (int ni = 0; ni < 4; ++ni) {
    int col = tileN + wc * 64 + ni * 16 + ln;
    cvalid[ni] = (col < Vsz);
    bcol[ni] = bias[cvalid[ni] ? col : (Vsz - 1)];
  }

  v8f acc[2][4];
#pragma unroll
  for (int mi = 0; mi < 2; ++mi)
#pragma unroll
    for (int ni = 0; ni < 4; ++ni)
#pragma unroll
      for (int e = 0; e < 8; ++e) acc[mi][ni][e] = 0.0f;

  const int rowA0 = tileM + wr * 32 + ln;
  for (int kb = 0; kb < KB32; ++kb) {
    v16bf afr[2], bfr[4];
#pragma unroll
    for (int mi = 0; mi < 2; ++mi) {
      size_t off = (((size_t)(rowA0 + mi * 16) * KB32 + kb) * 2 + half) * 16;
      afr[mi] = *reinterpret_cast<const v16bf*>(A + off);
    }
#pragma unroll
    for (int ni = 0; ni < 4; ++ni) {
      int col = tileN + wc * 64 + ni * 16 + ln;
      size_t off = ((size_t)(kb * 2 + half) * NPAD + col) * 16;
      bfr[ni] = *reinterpret_cast<const v16bf*>(Bw + off);
    }
#pragma unroll
    for (int mi = 0; mi < 2; ++mi)
#pragma unroll
      for (int ni = 0; ni < 4; ++ni)
        acc[mi][ni] = __builtin_amdgcn_wmma_f32_16x16x32_bf16(
            false, afr[mi], false, bfr[ni], (short)0, acc[mi][ni], false, false);
  }

  __shared__ float smax[128][2];
  __shared__ float ssum[128][2];

  // C layout: lane col = ln, row m = half*8 + r (r = accumulator element)
#pragma unroll
  for (int mi = 0; mi < 2; ++mi) {
#pragma unroll
    for (int r = 0; r < 8; ++r) {
      float vals[4];
      float rmax = NEGF;
#pragma unroll
      for (int ni = 0; ni < 4; ++ni) {
        float x = cvalid[ni] ? (acc[mi][ni][r] + bcol[ni]) : NEGF;
        vals[ni] = x;
        rmax = fmaxf(rmax, x);
      }
      for (int msk = 1; msk < 16; msk <<= 1)
        rmax = fmaxf(rmax, __shfl_xor(rmax, msk, 32));
      float s = 0.0f;
#pragma unroll
      for (int ni = 0; ni < 4; ++ni) s += __expf(vals[ni] - rmax);
      for (int msk = 1; msk < 16; msk <<= 1) s += __shfl_xor(s, msk, 32);
      if (ln == 0) {
        int rloc = wr * 32 + mi * 16 + half * 8 + r;
        smax[rloc][wc] = rmax;
        ssum[rloc][wc] = s;
      }
    }
  }
  __syncthreads();
  if (tid < 128) {
    float m0 = smax[tid][0], m1 = smax[tid][1];
    float s0 = ssum[tid][0], s1 = ssum[tid][1];
    float M = fmaxf(m0, m1);
    float S = __expf(m0 - M) * s0 + __expf(m1 - M) * s1;
    size_t o = (size_t)blockIdx.y * Mrows + tileM + tid;
    pmax[o] = M;
    psum[o] = S;
  }
}

// ---------------------------------------------------------------------------
// K3: fold 40 column-tile partials -> per-row logsumexp
// ---------------------------------------------------------------------------
__global__ void k_lse(const float* __restrict__ pmax, const float* __restrict__ psum,
                      float* __restrict__ lse) {
  int row = blockIdx.x * 256 + threadIdx.x;
  if (row >= Mrows) return;
  float m = NEGF, s = 0.0f;
  for (int ct = 0; ct < NCT; ++ct) {
    float pm = pmax[(size_t)ct * Mrows + row];
    float ps = psum[(size_t)ct * Mrows + row];
    if (pm > m) { s = s * __expf(m - pm) + ps; m = pm; }
    else        { s += ps * __expf(pm - m); }
  }
  lse[row] = m + __logf(s);
}

// ---------------------------------------------------------------------------
// K4a: gather bf16 B-fragments for label tokens per batch.
// lab[0]=blank(0), lab[1+j]=ys[b,j]; padded to 112 cols with zeros.
// ---------------------------------------------------------------------------
__global__ void k_blab(const unsigned short* __restrict__ Bw, const int* __restrict__ ys,
                       unsigned short* __restrict__ Bl) {
  int idx = blockIdx.x * 256 + threadIdx.x;          // (bb*32 + kb16)*NLAB + n
  if (idx >= Bsz * 32 * NLAB) return;
  int n    = idx % NLAB;
  int kb16 = (idx / NLAB) % 32;
  int bb   = idx / (NLAB * 32);
  unsigned short* dst = Bl + (size_t)idx * 16;
  if (n <= Lsz) {
    int v = (n == 0) ? 0 : ys[bb * Lsz + n - 1];
    const unsigned short* src = Bw + ((size_t)kb16 * NPAD + v) * 16;
#pragma unroll
    for (int j = 0; j < 16; ++j) dst[j] = src[j];
  } else {
#pragma unroll
    for (int j = 0; j < 16; ++j) dst[j] = 0;
  }
}

// ---------------------------------------------------------------------------
// K4b: per-batch label GEMM [800x512] x [512x112] (bf16 WMMA),
// epilogue fuses +bias[label] - lse -> log-prob of each label token.
// Grid: (7 row tiles of 128, 32 batches); wave = 16 rows x 112 cols.
// Label/bias gather hoisted branchlessly before the K loop.
// ---------------------------------------------------------------------------
__global__ __launch_bounds__(256) void k_labgemm(
    const unsigned short* __restrict__ A, const unsigned short* __restrict__ Blab,
    const int* __restrict__ ys, const float* __restrict__ bias,
    const float* __restrict__ lse, float* __restrict__ lablp) {
  const int tid  = threadIdx.x;
  const int wave = tid >> 5, lane = tid & 31;
  const int half = lane >> 4, ln = lane & 15;
  const int bb    = blockIdx.y;
  const int rbase = blockIdx.x * 128 + wave * 16;    // t base (may exceed 799)

  // branchless label + bias preload (latency hidden under K loop)
  bool  valid[7];
  float bv[7];
#pragma unroll
  for (int ni = 0; ni < 7; ++ni) {
    int n = ni * 16 + ln;
    valid[ni] = (n <= Lsz);
    int nc = valid[ni] ? n : Lsz;                    // clamp
    int vlab = (nc == 0) ? 0 : ys[bb * Lsz + nc - 1];
    bv[ni] = bias[vlab];
  }

  v8f acc[7];
#pragma unroll
  for (int ni = 0; ni < 7; ++ni)
#pragma unroll
    for (int e = 0; e < 8; ++e) acc[ni][e] = 0.0f;

  const unsigned short* Bl = Blab + (size_t)bb * (32 * NLAB * 16);
  int tA = rbase + ln;
  if (tA > Tsz - 1) tA = Tsz - 1;                    // clamp tail-tile loads
  size_t arow = (size_t)(bb * Tsz + tA);

  for (int kb = 0; kb < KB32; ++kb) {
    size_t aoff = ((arow * KB32 + kb) * 2 + half) * 16;
    v16bf a = *reinterpret_cast<const v16bf*>(A + aoff);
#pragma unroll
    for (int ni = 0; ni < 7; ++ni) {
      size_t boff = ((size_t)(kb * 2 + half) * NLAB + ni * 16 + ln) * 16;
      v16bf b = *reinterpret_cast<const v16bf*>(Bl + boff);
      acc[ni] = __builtin_amdgcn_wmma_f32_16x16x32_bf16(
          false, a, false, b, (short)0, acc[ni], false, false);
    }
  }

#pragma unroll
  for (int ni = 0; ni < 7; ++ni) {
    int n = ni * 16 + ln;
#pragma unroll
    for (int r = 0; r < 8; ++r) {
      int t = rbase + half * 8 + r;
      if (t < Tsz) {
        float x = valid[ni] ? (acc[ni][r] + bv[ni] - lse[bb * Tsz + t]) : NEGF;
        lablp[((size_t)bb * Tsz + t) * NLAB + n] = x;
      }
    }
  }
}

// ---------------------------------------------------------------------------
// K5: CTC alpha recursion, one block per batch, thread per extended state.
// lablp[b][t][j]: j=0 blank, j=1+l label l.  loss accumulated into out.
// ---------------------------------------------------------------------------
__global__ __launch_bounds__(256) void k_ctc(
    const float* __restrict__ lablp, const int* __restrict__ ys,
    const int* __restrict__ ys_lens, const int* __restrict__ hlens,
    float* __restrict__ out) {
  __shared__ float al[2][224];
  const int bb = blockIdx.x;
  const int s  = threadIdx.x;
  const int Lb = ys_lens[bb];
  const int hlen = hlens[bb];
  const int S = 2 * Lb + 1;
  const float* lp = lablp + (size_t)bb * Tsz * NLAB;

  int li = (s & 1) ? (1 + (s >> 1)) : 0;             // lablp column for state s
  bool allow2 = false;
  if ((s & 1) && s >= 3 && s < S)
    allow2 = (ys[bb * Lsz + (s >> 1)] != ys[bb * Lsz + (s >> 1) - 1]);

  if (s < 224) {
    float v = NEGF;
    if (s == 0) v = lp[0];
    else if (s == 1 && Lb > 0) v = lp[li];
    al[0][s] = v;
  }
  __syncthreads();

  for (int t = 1; t < Tsz; ++t) {
    int cur = t & 1, prev = cur ^ 1;
    if (s < S) {
      float a0 = al[prev][s];
      float a1 = (s >= 1) ? al[prev][s - 1] : NEGF;
      float a2 = allow2 ? al[prev][s - 2] : NEGF;
      float m = fmaxf(a0, fmaxf(a1, a2));
      float nv = lp[(size_t)t * NLAB + li] + m +
                 __logf(__expf(a0 - m) + __expf(a1 - m) + __expf(a2 - m));
      if (t >= hlen) nv = a0;                         // freeze past input length
      al[cur][s] = nv;
    }
    __syncthreads();
  }

  if (s == 0) {
    int lb = (Tsz - 1) & 1;
    float aL = al[lb][2 * Lb];
    float aP = (Lb > 0) ? al[lb][2 * Lb - 1] : NEGF;
    float m = fmaxf(aL, aP);
    float loss = -(m + __logf(__expf(aL - m) + __expf(aP - m)));
    atomicAdd(out, loss / (float)Bsz);
  }
}

// ---------------------------------------------------------------------------
// Host launcher
// ---------------------------------------------------------------------------
extern "C" void kernel_launch(void* const* d_in, const int* in_sizes, int n_in,
                              void* d_out, int out_size, void* d_ws, size_t ws_size,
                              hipStream_t stream) {
  (void)in_sizes; (void)n_in; (void)out_size; (void)ws_size;
  const float* hs      = (const float*)d_in[0];
  const int*   hlens   = (const int*)  d_in[1];
  const int*   ys      = (const int*)  d_in[2];
  const int*   ys_lens = (const int*)  d_in[3];
  const float* W       = (const float*)d_in[4];
  const float* bias    = (const float*)d_in[5];
  float* out = (float*)d_out;

  char* ws = (char*)d_ws;
  // 256B-aligned workspace layout (total ~54.9 MB)
  unsigned short* Aswz = (unsigned short*)(ws + 0);          // 26,214,400 B
  unsigned short* Bswz = (unsigned short*)(ws + 26214400);   //  5,242,880 B
  float* pmax          = (float*)(ws + 31457280);            //  4,096,000 B
  float* psum          = (float*)(ws + 35553280);            //  4,096,000 B
  float* lse           = (float*)(ws + 39649280);            //    102,400 B
  unsigned short* Blab = (unsigned short*)(ws + 39751680);   //  3,670,016 B
  float* lablp         = (float*)(ws + 43421696);            // 11,468,800 B

  k_zero<<<1, 64, 0, stream>>>(out);
  k_convA<<<(Mrows * KB32 * 2 + 255) / 256, 256, 0, stream>>>(hs, Aswz);
  k_convW<<<(32 * NPAD + 255) / 256, 256, 0, stream>>>(W, Bswz);
  k_gemm<<<dim3(Mrows / 128, NCT), 256, 0, stream>>>(Aswz, Bswz, bias, pmax, psum);
  k_lse<<<(Mrows + 255) / 256, 256, 0, stream>>>(pmax, psum, lse);
  k_blab<<<(Bsz * 32 * NLAB + 255) / 256, 256, 0, stream>>>(Bswz, ys, Blab);
  k_labgemm<<<dim3(7, Bsz), 256, 0, stream>>>(Aswz, Blab, ys, bias, lse, lablp);
  k_ctc<<<Bsz, 256, 0, stream>>>(lablp, ys, ys_lens, hlens, out);
}
